// DragonHGT_85538568667161
// MI455X (gfx1250) — compile-verified, matrix operand order checked by reading, not run on hardware
//
#include <hip/hip_runtime.h>
#include <math.h>

// ---------------- CDNA5 WMMA types ----------------
typedef __attribute__((ext_vector_type(16))) __bf16 v16bf;
typedef __attribute__((ext_vector_type(8)))  __bf16 v8bf;
typedef __attribute__((ext_vector_type(8)))  float  v8f;

// order-preserving float<->uint encoding for atomicMax-based segment max
__device__ __forceinline__ unsigned fenc(float f) {
  unsigned u = __float_as_uint(f);
  return (u & 0x80000000u) ? ~u : (u | 0x80000000u);
}
__device__ __forceinline__ float fdec(unsigned e) {
  unsigned u = (e & 0x80000000u) ? (e & 0x7FFFFFFFu) : ~e;
  return __uint_as_float(u);
}

// async global->LDS copy of 16B per lane (CDNA5, ASYNCcnt-tracked)
__device__ __forceinline__ void async_ld_b128(unsigned lds_byte_addr, const float* gaddr) {
  asm volatile("global_load_async_to_lds_b128 %0, %1, off"
               :: "v"(lds_byte_addr), "v"(gaddr) : "memory");
}
__device__ __forceinline__ void wait_async0() {
  asm volatile("s_wait_asynccnt 0x0" ::: "memory");
}

// build A fragment (bf16 16x32 per ISA layout) from f32 LDS row.
// element i  <->  K = (i>>3)*16 + khalf*8 + (i&7); casts lower to HW bf16 cvt.
__device__ __forceinline__ v16bf build_a(const float* row, int khalf) {
  float4 a0 = *(const float4*)(row + khalf * 8);
  float4 a1 = *(const float4*)(row + khalf * 8 + 4);
  float4 a2 = *(const float4*)(row + 16 + khalf * 8);
  float4 a3 = *(const float4*)(row + 16 + khalf * 8 + 4);
  v16bf f;
  f[0]  = (__bf16)a0.x; f[1]  = (__bf16)a0.y; f[2]  = (__bf16)a0.z; f[3]  = (__bf16)a0.w;
  f[4]  = (__bf16)a1.x; f[5]  = (__bf16)a1.y; f[6]  = (__bf16)a1.z; f[7]  = (__bf16)a1.w;
  f[8]  = (__bf16)a2.x; f[9]  = (__bf16)a2.y; f[10] = (__bf16)a2.z; f[11] = (__bf16)a2.w;
  f[12] = (__bf16)a3.x; f[13] = (__bf16)a3.y; f[14] = (__bf16)a3.z; f[15] = (__bf16)a3.w;
  return f;
}
// build B fragment from bf16 transposed LDS row (column of W)
__device__ __forceinline__ v16bf build_b(const __bf16* row, int khalf) {
  union { v16bf v; v8bf h[2]; } f;
  f.h[0] = *(const v8bf*)(row + khalf * 8);
  f.h[1] = *(const v8bf*)(row + 16 + khalf * 8);
  return f.v;
}

// ---------------- bf16 WMMA GEMM: C = act(A @ W + bias) ----------------
// Block tile 128x64, 256 threads = 8 waves in 4x2; each wave does a 32x32
// output patch = 2x2 v_wmma_f32_16x16x32_bf16 per 32-deep K step.
// Requirements: K % 32 == 0, N % 64 == 0 (true at all call sites);
// M edge handled by clamping loads + guarding stores.
#define TM 128
#define TN 64
#define TK 32
#define A_STRIDE 36   // f32; 144B row pitch (16B multiple, bank-staggered)
#define B_STRIDE 40   // bf16; 80B row pitch (16B multiple, bank-staggered)

__global__ __launch_bounds__(256)
void gemm_wmma(const float* __restrict__ A, long sAm, long sAb,
               const float* __restrict__ W, long sWk, long sWb,
               const float* __restrict__ bias,
               float* __restrict__ C, long sCm, long sCb,
               int M, int K, int N, int act) {
  __shared__ float  As[TM][A_STRIDE];   // raw f32 A tile (async-staged)
  __shared__ __bf16 Bs[TN][B_STRIDE];   // bf16 W tile, TRANSPOSED [n][k]
  const int tid = threadIdx.x;
  const int bM  = blockIdx.x * TM;
  const int bN  = blockIdx.y * TN;
  A += (long)blockIdx.z * sAb;
  W += (long)blockIdx.z * sWb;
  C += (long)blockIdx.z * sCb;

  const int wid   = tid >> 5;
  const int lane  = tid & 31;
  const int wm    = wid >> 1;     // 0..3
  const int wn    = wid & 1;      // 0..1
  const int khalf = lane >> 4;    // 0/1
  const int l15   = lane & 15;

  const unsigned ldsA = (unsigned)(size_t)(&As[0][0]);
  const int ar = (lane >> 3);          // 0..3 row within group of 4
  const int ac = (lane & 7) * 4;       // col group of 4 f32
  const int kb = tid & 31;             // B staging: W row k0+kb
  const int nb = (tid >> 5) * 8;       // 8 n's per thread

  v8f acc00 = {}, acc01 = {}, acc10 = {}, acc11 = {};

  for (int k0 = 0; k0 < K; k0 += TK) {
    // --- stage A (128x32 f32) via async global->LDS, rows clamped to M-1 ---
#pragma unroll
    for (int i = 0; i < 4; ++i) {
      int r = wid * 16 + i * 4 + ar;
      int gm = bM + r; if (gm > M - 1) gm = M - 1;
      async_ld_b128(ldsA + (unsigned)(r * A_STRIDE + ac) * 4u,
                    A + (long)gm * sAm + k0 + ac);
    }
    // --- stage B (32x64) -> bf16 transposed (HW cvt), vectorized reads ---
    {
      const float* src = W + (long)(k0 + kb) * sWk + bN + nb;
      float4 w0 = *(const float4*)src;
      float4 w1 = *(const float4*)(src + 4);
      Bs[nb + 0][kb] = (__bf16)w0.x;
      Bs[nb + 1][kb] = (__bf16)w0.y;
      Bs[nb + 2][kb] = (__bf16)w0.z;
      Bs[nb + 3][kb] = (__bf16)w0.w;
      Bs[nb + 4][kb] = (__bf16)w1.x;
      Bs[nb + 5][kb] = (__bf16)w1.y;
      Bs[nb + 6][kb] = (__bf16)w1.z;
      Bs[nb + 7][kb] = (__bf16)w1.w;
    }
    wait_async0();
    __syncthreads();

    // --- fragments + 2x2 WMMA ---
    v16bf fa0 = build_a(&As[wm * 32 + l15][0], khalf);
    v16bf fa1 = build_a(&As[wm * 32 + 16 + l15][0], khalf);
    v16bf fb0 = build_b(&Bs[wn * 32 + l15][0], khalf);
    v16bf fb1 = build_b(&Bs[wn * 32 + 16 + l15][0], khalf);
    acc00 = __builtin_amdgcn_wmma_f32_16x16x32_bf16(false, fa0, false, fb0, (short)0, acc00, false, false);
    acc01 = __builtin_amdgcn_wmma_f32_16x16x32_bf16(false, fa0, false, fb1, (short)0, acc01, false, false);
    acc10 = __builtin_amdgcn_wmma_f32_16x16x32_bf16(false, fa1, false, fb0, (short)0, acc10, false, false);
    acc11 = __builtin_amdgcn_wmma_f32_16x16x32_bf16(false, fa1, false, fb1, (short)0, acc11, false, false);
    __syncthreads();
  }

  // --- store: D layout VGPR r -> m = 16*mt + 8*khalf + r, n = 16*nt + l15 ---
  const v8f* accs[2][2] = {{&acc00, &acc01}, {&acc10, &acc11}};
#pragma unroll
  for (int mt = 0; mt < 2; ++mt) {
#pragma unroll
    for (int nt = 0; nt < 2; ++nt) {
      int n = bN + wn * 32 + nt * 16 + l15;
      if (n >= N) continue;
      float bv = bias ? bias[n] : 0.0f;
      const v8f a = *accs[mt][nt];
#pragma unroll
      for (int r = 0; r < 8; ++r) {
        int m = bM + wm * 32 + mt * 16 + khalf * 8 + r;
        if (m < M) {
          float v = a[r] + bv;
          if (act == 1) v = fmaxf(v, 0.0f);
          C[(long)m * sCm + n] = v;
        }
      }
    }
  }
}

// ---------------- attention: alpha + segment max ----------------
__global__ __launch_bounds__(256)
void edge_alpha(const int* __restrict__ ei, int E,
                const float* __restrict__ Q, const float* __restrict__ KS,
                const float* __restrict__ p_rel,
                float* __restrict__ alpha, unsigned* __restrict__ amax) {
  int e = blockIdx.x;
  int h = threadIdx.x >> 5;
  int lane = threadIdx.x & 31;
  int src = ei[e], dst = ei[E + e];
  const float4* q4 = (const float4*)(Q + (long)dst * 1024 + h * 128);
  const float4* k4 = (const float4*)(KS + (long)src * 1024 + h * 128);
  float4 a = q4[lane], b = k4[lane];
  float s = a.x * b.x + a.y * b.y + a.z * b.z + a.w * b.w;
#pragma unroll
  for (int o = 16; o; o >>= 1) s += __shfl_xor(s, o);
  if (lane == 0) {
    float al = s * p_rel[h] * 0.08838834764831845f;  // 1/sqrt(128)
    alpha[(long)e * 8 + h] = al;
    atomicMax(&amax[(long)dst * 8 + h], fenc(al));
  }
}

__global__ __launch_bounds__(256)
void edge_accum(const int* __restrict__ ei, int E,
                const float* __restrict__ VS, const float* __restrict__ alpha,
                const unsigned* __restrict__ amax,
                float* __restrict__ den, float* __restrict__ num) {
  int e = blockIdx.x;
  int h = threadIdx.x >> 5;
  int lane = threadIdx.x & 31;
  int src = ei[e], dst = ei[E + e];
  float al = alpha[(long)e * 8 + h];
  float mx = fdec(amax[(long)dst * 8 + h]);
  float ex = __expf(al - mx);
  if (lane == 0) atomicAdd(&den[(long)dst * 8 + h], ex);
  const float4* v4 = (const float4*)(VS + (long)src * 1024 + h * 128);
  float4 v = v4[lane];
  float* nr = num + (long)dst * 1024 + h * 128 + lane * 4;
  atomicAdd(nr + 0, ex * v.x);
  atomicAdd(nr + 1, ex * v.y);
  atomicAdd(nr + 2, ex * v.z);
  atomicAdd(nr + 3, ex * v.w);
}

__global__ void norm_add(float* __restrict__ agg, const float* __restrict__ num,
                         const float* __restrict__ den, long total) {
  long i = (long)blockIdx.x * 256 + threadIdx.x;
  if (i >= total) return;
  long n = i >> 10;
  int h = (int)((i >> 7) & 7);
  agg[i] += num[i] / (den[n * 8 + h] + 1e-16f);
}

// ---------------- elementwise ----------------
__global__ void fill_f32(float* p, long n, float v) {
  long i = (long)blockIdx.x * 256 + threadIdx.x;
  if (i < n) p[i] = v;
}
__global__ void fill_u32(unsigned* p, long n, unsigned v) {
  long i = (long)blockIdx.x * 256 + threadIdx.x;
  if (i < n) p[i] = v;
}
__global__ void gelu_ip(float* p, long n) {
  long i = (long)blockIdx.x * 256 + threadIdx.x;
  if (i >= n) return;
  float x = p[i];
  p[i] = 0.5f * x * (1.0f + erff(x * 0.70710678118654752f));
}
__global__ void skip_relu(float* __restrict__ cur, const float* __restrict__ r,
                          const float* __restrict__ skipp, int do_relu, long n) {
  long i = (long)blockIdx.x * 256 + threadIdx.x;
  if (i >= n) return;
  float v = r[i];
  if (skipp) {
    float sk = 1.0f / (1.0f + __expf(-skipp[0]));
    v = sk * v + (1.0f - sk) * cur[i];
  }
  if (do_relu) v = fmaxf(v, 0.0f);
  cur[i] = v;
}

// ---------------- readout ----------------
__global__ void searchsorted64(const int* __restrict__ batch, int n, int* __restrict__ n0) {
  int g = threadIdx.x;
  if (g >= 64) return;
  int lo = 0, hi = n;
  while (lo < hi) { int mid = (lo + hi) >> 1; if (batch[mid] < g) lo = mid + 1; else hi = mid; }
  n0[g] = lo;
}
__global__ void gather_rows(const float* __restrict__ X, const int* __restrict__ idx,
                            float* __restrict__ out) {
  int g = blockIdx.x;
  int r = idx[g];
  for (int j = threadIdx.x; j < 1024; j += 256) out[(long)g * 1024 + j] = X[(long)r * 1024 + j];
}
__global__ void head_final(const float* __restrict__ H, const float* __restrict__ W,
                           const float* __restrict__ B, float* __restrict__ out, int od) {
  int t = threadIdx.x;
  int g = t / od, o = t % od;
  if (g >= 64) return;
  float s = B[o];
  for (int k = 0; k < 512; ++k) s += H[g * 512 + k] * W[k * od + o];
  out[g * od + o] = s;
}

// ---------------- host orchestration ----------------
static inline long cdiv(long a, long b) { return (a + b - 1) / b; }

extern "C" void kernel_launch(void* const* d_in, const int* in_sizes, int n_in,
                              void* d_out, int out_size, void* d_ws, size_t ws_size,
                              hipStream_t stream) {
  (void)in_sizes; (void)n_in; (void)out_size; (void)ws_size;
#define IN(i) ((const float*)d_in[(i)])
#define INI(i) ((const int*)d_in[(i)])
  const int ND = 10000, NE = 20000, NS = 10000, NT = 40000, EC = 50000;
  const int NN[3] = {ND, NE, NS};
  const long xoff[3] = {0L, (long)ND * 1024, (long)(ND + NE) * 1024};
  // edge types (sorted key order): D->E, E->D, E->E, E->S, S->E
  const int est[5] = {0, 1, 1, 1, 2};
  const int edt[5] = {1, 0, 1, 2, 1};

  // workspace carve-up (floats)
  float* w = (float*)d_ws;
  long off = 0;
  float* CUR  = w + off; off += (long)NT * 1024;
  float* KB   = w + off; off += (long)NT * 1024;
  float* QB   = w + off; off += (long)NT * 1024;
  float* VB   = w + off; off += (long)NT * 1024;
  float* AGG  = w + off; off += (long)NT * 1024;
  float* RB   = w + off; off += (long)NT * 1024;
  float* KS   = w + off; off += (long)NE * 1024;
  float* VS   = w + off; off += (long)NE * 1024;
  float* NUM  = w + off; off += (long)NE * 1024;
  float* DEN  = w + off; off += (long)NE * 8;
  unsigned* AMAX = (unsigned*)(w + off); off += (long)NE * 8;
  float* ALPHA = w + off; off += (long)EC * 8;
  float* HA = w + off; off += 64 * 1024;
  float* HB = w + off; off += 64 * 1024;
  float* HS = w + off; off += 64 * 512;
  int* N0 = (int*)(w + off); off += 64;

  const float* xin[3] = {IN(208), IN(209), IN(210)};

  auto gemm = [&](const float* A, long sAm, long sAb, const float* W, long sWk, long sWb,
                  const float* bias, float* C, long sCm, long sCb,
                  int M, int K, int N, int Z, int act) {
    dim3 g((unsigned)cdiv(M, TM), (unsigned)cdiv(N, TN), (unsigned)Z);
    hipLaunchKernelGGL(gemm_wmma, g, dim3(256), 0, stream,
                       A, sAm, sAb, W, sWk, sWb, bias, C, sCm, sCb, M, K, N, act);
  };
  auto fillf = [&](float* p, long n, float v) {
    hipLaunchKernelGGL(fill_f32, dim3((unsigned)cdiv(n, 256)), dim3(256), 0, stream, p, n, v);
  };
  auto fillu = [&](unsigned* p, long n, unsigned v) {
    hipLaunchKernelGGL(fill_u32, dim3((unsigned)cdiv(n, 256)), dim3(256), 0, stream, p, n, v);
  };

  for (int l = 0; l < 4; ++l) {
    const int base = 6 + 42 * l;
    const int K = (l == 0) ? 256 : 1024;
    // K/Q/V projections (bf16 WMMA)
    for (int t = 0; t < 3; ++t) {
      const float* A = (l == 0) ? xin[t] : (CUR + xoff[t]);
      gemm(A, K, 0, IN(base + 12 + 2 * t), 1024, 0, IN(base + 11 + 2 * t),
           KB + xoff[t], 1024, 0, NN[t], K, 1024, 1, 0);
      gemm(A, K, 0, IN(base + 28 + 2 * t), 1024, 0, IN(base + 27 + 2 * t),
           QB + xoff[t], 1024, 0, NN[t], K, 1024, 1, 0);
      gemm(A, K, 0, IN(base + 37 + 2 * t), 1024, 0, IN(base + 36 + 2 * t),
           VB + xoff[t], 1024, 0, NN[t], K, 1024, 1, 0);
    }
    fillf(AGG, (long)NT * 1024, 0.0f);
    // per edge type: relation einsums (8-way batched 128x128 WMMA) + attention
    for (int et = 0; et < 5; ++et) {
      const int s = est[et], d = edt[et];
      gemm(KB + xoff[s], 1024, 128, IN(base + 6 + et), 128, 16384, nullptr,
           KS, 1024, 128, NN[s], 128, 128, 8, 0);
      gemm(VB + xoff[s], 1024, 128, IN(base + 17 + et), 128, 16384, nullptr,
           VS, 1024, 128, NN[s], 128, 128, 8, 0);
      fillu(AMAX, (long)NN[d] * 8, 0x007FFFFFu);   // fenc(-inf)
      fillf(DEN, (long)NN[d] * 8, 0.0f);
      fillf(NUM, (long)NN[d] * 1024, 0.0f);
      hipLaunchKernelGGL(edge_alpha, dim3(EC), dim3(256), 0, stream,
                         INI(1 + et), EC, QB + xoff[d], KS, IN(base + 22 + et), ALPHA, AMAX);
      hipLaunchKernelGGL(edge_accum, dim3(EC), dim3(256), 0, stream,
                         INI(1 + et), EC, VS, ALPHA, AMAX, DEN, NUM);
      long tot = (long)NN[d] * 1024;
      hipLaunchKernelGGL(norm_add, dim3((unsigned)cdiv(tot, 256)), dim3(256), 0, stream,
                         AGG + xoff[d], NUM, DEN, tot);
    }
    // gelu -> output projection -> skip/relu
    for (int t = 0; t < 3; ++t) {
      long n = (long)NN[t] * 1024;
      hipLaunchKernelGGL(gelu_ip, dim3((unsigned)cdiv(n, 256)), dim3(256), 0, stream,
                         AGG + xoff[t], n);
      gemm(AGG + xoff[t], 1024, 0, IN(base + 2 * t + 1), 1024, 0, IN(base + 2 * t),
           RB + xoff[t], 1024, 0, NN[t], 1024, 1024, 1, 0);
      hipLaunchKernelGGL(skip_relu, dim3((unsigned)cdiv(n, 256)), dim3(256), 0, stream,
                         CUR + xoff[t], RB + xoff[t],
                         (l > 0) ? IN(base + 33 + t) : (const float*)nullptr,
                         (l < 3) ? 1 : 0, n);
    }
  }

  // readout
  hipLaunchKernelGGL(searchsorted64, dim3(1), dim3(64), 0, stream, INI(0), ND, N0);
  hipLaunchKernelGGL(gather_rows, dim3(64), dim3(256), 0, stream, CUR, N0, HA);
  gemm(HA, 1024, 0, IN(203), 1024, 0, IN(202), HB, 1024, 0, 64, 1024, 1024, 1, 1);
  gemm(HB, 1024, 0, IN(205), 1024, 0, IN(204), HA, 1024, 0, 64, 1024, 1024, 1, 1);
  gemm(HA, 1024, 0, IN(207), 1024, 0, IN(206), HB, 1024, 0, 64, 1024, 1024, 1, 1);
  struct HeadDesc { int p; int od; int ooff; };
  const HeadDesc hd[7] = {
    {174 + 4 * 4, 3, 0},     // ptr_l1
    {174 + 4 * 5, 3, 192},   // ptr_l2
    {174 + 4 * 6, 3, 384},   // ptr_l3
    {174 + 4 * 0, 5, 576},   // leaf_category
    {174 + 4 * 2, 1, 896},   // leaf_signed
    {174 + 4 * 1, 1, 960},   // leaf_floating
    {174 + 4 * 3, 7, 1024},  // leaf_size
  };
  for (int i = 0; i < 7; ++i) {
    gemm(HB, 1024, 0, IN(hd[i].p + 3), 512, 0, IN(hd[i].p + 2),
         HS, 512, 0, 64, 1024, 512, 1, 1);
    hipLaunchKernelGGL(head_final, dim3(1), dim3(64 * hd[i].od), 0, stream,
                       HS, IN(hd[i].p + 1), IN(hd[i].p + 0),
                       (float*)d_out + hd[i].ooff, hd[i].od);
  }
#undef IN
#undef INI
}
